// MultiHeadAttention_33913061769280
// MI455X (gfx1250) — matrix operand order repaired and checked by reading, not compiled, not run on hardware
//
#include <hip/hip_runtime.h>

// ---------------------------------------------------------------------------
// MHA forward for MI455X (gfx1250, wave32, WMMA bf16 16x16x32)
// B=8, T=1024, D=768, H=12, Dh=64
// v3: K-step 64 staged tiles, explicit B-fragment regs, exp2-domain softmax.
// ---------------------------------------------------------------------------

#define Bsz   8
#define Tsz   1024
#define Dsz   768
#define Hsz   12
#define DHsz  64
#define D3sz  2304
#define Msz   (Bsz * Tsz)                 // 8192 rows
#define QSCALE 0.18033688011112042f       // (1/sqrt(64)) * log2(e); softmax in exp2 domain

typedef __attribute__((ext_vector_type(16))) __bf16        v16bf;
typedef __attribute__((ext_vector_type(8)))  float         v8f;
typedef __attribute__((ext_vector_type(4)))  unsigned int  u32x4;

union FragU {
    u32x4 u[2];
    v16bf v;
};

// -------------------------- async copy primitives --------------------------
__device__ __forceinline__ unsigned int lds_addr_of(const void* p) {
    return (unsigned int)(size_t)p;       // LDS aperture: offset in addr[31:0]
}

__device__ __forceinline__ void async_b128(unsigned int lds_off, const void* g) {
    asm volatile("global_load_async_to_lds_b128 %0, %1, off"
                 :: "v"(lds_off), "v"(g) : "memory");
}

#define ASYNC_WAIT_4() asm volatile("s_wait_asynccnt 0x4" ::: "memory")
#define ASYNC_WAIT_8() asm volatile("s_wait_asynccnt 0x8" ::: "memory")

// -------------------------- WMMA fragment loaders --------------------------
// A-matrix fragment (16x32, bf16), row-major src, leading dim `ld`.
__device__ __forceinline__ v16bf load_frag_a(const __bf16* src, int ld, int row0, int k0) {
    int lane = threadIdx.x & 31;
    int hf   = lane >> 4;
    int m    = lane & 15;
    const __bf16* p = src + (size_t)(row0 + m) * ld + k0 + hf * 8;
    FragU f;
    f.u[0] = *(const u32x4*)p;
    f.u[1] = *(const u32x4*)(p + 16);
    return f.v;
}

// B-matrix fragment (32x16, bf16) with B[k][n] = src[n0+n][k0+k].
__device__ __forceinline__ v16bf load_frag_bt(const __bf16* src, int ld, int n0, int k0) {
    int lane = threadIdx.x & 31;
    int hf   = lane >> 4;
    int n    = lane & 15;
    const __bf16* p = src + (size_t)(n0 + n) * ld + k0 + hf * 8;
    FragU f;
    f.u[0] = *(const u32x4*)p;
    f.u[1] = *(const u32x4*)(p + 16);
    return f.v;
}

__device__ __forceinline__ v8f wmma_bf16(v16bf a, v16bf b, v8f c) {
    return __builtin_amdgcn_wmma_f32_16x16x32_bf16(
        false, a, false, b, (short)0, c, false, false);
}

// ---------------------------------------------------------------------------
// Kernel 1: f32 -> bf16 conversion (grid-stride)
// ---------------------------------------------------------------------------
__global__ void cvt_f32_to_bf16(const float* __restrict__ in, __bf16* __restrict__ out, int n) {
    int i = blockIdx.x * blockDim.x + threadIdx.x;
    int stride = gridDim.x * blockDim.x;
    for (; i < n; i += stride) out[i] = (__bf16)in[i];
}

// ---------------------------------------------------------------------------
// GEMM mainloop: block = 128(M)x128(N), K=768 in 64-wide staged steps.
// Waves 4(M) x 2(N); each wave 32x64 -> acc[2][4].
// A/B 128x64 bf16 tiles (16KB each) double-buffered via async loads.
// ---------------------------------------------------------------------------
__device__ __forceinline__ void stage_tile_128x64(const __bf16* __restrict__ g, int ld,
                                                  int row0, int k0, __bf16* lbuf) {
    int tid = threadIdx.x;                 // 256 threads x 64B
    int row = tid >> 1;
    int hc  = (tid & 1) * 32;              // 32-elem (64B) half-row
    const __bf16* gp = g + (size_t)(row0 + row) * ld + k0 + hc;
    __bf16* lp = lbuf + row * 64 + hc;
    unsigned int l = lds_addr_of(lp);
    async_b128(l,      gp);
    async_b128(l + 16, gp + 8);
    async_b128(l + 32, gp + 16);
    async_b128(l + 48, gp + 24);
}

__device__ __forceinline__ void gemm_mainloop_128(const __bf16* __restrict__ A,
                                                  const __bf16* __restrict__ Bm,
                                                  int m0_blk, int n0_blk,
                                                  __bf16* Ab, __bf16* Bb,
                                                  int wm, int wn, v8f acc[2][4]) {
    stage_tile_128x64(A,  Dsz, m0_blk, 0,  Ab);
    stage_tile_128x64(Bm, Dsz, n0_blk, 0,  Bb);
    stage_tile_128x64(A,  Dsz, m0_blk, 64, Ab + 8192);
    stage_tile_128x64(Bm, Dsz, n0_blk, 64, Bb + 8192);

    for (int it = 0; it < 12; ++it) {
        ASYNC_WAIT_8();                    // current buffer done; next still in flight
        __syncthreads();
        const __bf16* Ac = Ab + (it & 1) * 8192;
        const __bf16* Bc = Bb + (it & 1) * 8192;
#pragma unroll
        for (int kc = 0; kc < 64; kc += 32) {
            v16bf a0 = load_frag_a(Ac, 64, wm,      kc);
            v16bf a1 = load_frag_a(Ac, 64, wm + 16, kc);
            v16bf b0 = load_frag_bt(Bc, 64, wn,      kc);
            v16bf b1 = load_frag_bt(Bc, 64, wn + 16, kc);
            v16bf b2 = load_frag_bt(Bc, 64, wn + 32, kc);
            v16bf b3 = load_frag_bt(Bc, 64, wn + 48, kc);
            acc[0][0] = wmma_bf16(a0, b0, acc[0][0]);
            acc[1][0] = wmma_bf16(a1, b0, acc[1][0]);
            acc[0][1] = wmma_bf16(a0, b1, acc[0][1]);
            acc[1][1] = wmma_bf16(a1, b1, acc[1][1]);
            acc[0][2] = wmma_bf16(a0, b2, acc[0][2]);
            acc[1][2] = wmma_bf16(a1, b2, acc[1][2]);
            acc[0][3] = wmma_bf16(a0, b3, acc[0][3]);
            acc[1][3] = wmma_bf16(a1, b3, acc[1][3]);
        }
        __syncthreads();
        if (it + 2 < 12) {
            stage_tile_128x64(A,  Dsz, m0_blk, (it + 2) * 64, Ab + (it & 1) * 8192);
            stage_tile_128x64(Bm, Dsz, n0_blk, (it + 2) * 64, Bb + (it & 1) * 8192);
        }
    }
}

// ---------------------------------------------------------------------------
// Kernel 2: QKV projection + scatter to q (pre-scaled, exp2 domain),
// k [B,H,T,Dh] and vT [B,H,Dh,T].  Grid: 64 x 18 = 1152 blocks.
// ---------------------------------------------------------------------------
__global__ void qkv_gemm(const __bf16* __restrict__ xb, const __bf16* __restrict__ wq,
                         const float* __restrict__ bias,
                         __bf16* __restrict__ qb, __bf16* __restrict__ kb,
                         __bf16* __restrict__ vtb) {
    __shared__ __align__(16) __bf16 Ab[2 * 8192];
    __shared__ __align__(16) __bf16 Bb[2 * 8192];
    int mb = blockIdx.x & 63;
    int nb = blockIdx.x >> 6;
    int m0_blk = mb * 128, n0_blk = nb * 128;
    int wave = threadIdx.x >> 5;
    int wm = (wave >> 1) * 32;
    int wn = (wave & 1) * 64;
    int lane = threadIdx.x & 31;
    int hf = lane >> 4, col = lane & 15;

    v8f acc[2][4] = {};
    gemm_mainloop_128(xb, wq, m0_blk, n0_blk, Ab, Bb, wm, wn, acc);

#pragma unroll
    for (int nt = 0; nt < 4; ++nt) {
        int e  = n0_blk + wn + nt * 16 + col;
        float bv = bias[e];
        int s   = e / Dsz;
        int rem = e - s * Dsz;
        int h   = rem >> 6;
        int dh  = rem & 63;
#pragma unroll
        for (int mt = 0; mt < 2; ++mt) {
#pragma unroll
            for (int r = 0; r < 8; ++r) {
                int gm = m0_blk + wm + mt * 16 + r + 8 * hf;
                int bb = gm >> 10;
                int t  = gm & 1023;
                float v = acc[mt][nt][r] + bv;
                size_t head = (size_t)(bb * Hsz + h);
                if (s == 0) {
                    qb[(head * Tsz + t) * DHsz + dh] = (__bf16)(v * QSCALE);
                } else if (s == 1) {
                    kb[(head * Tsz + t) * DHsz + dh] = (__bf16)v;
                } else {
                    vtb[(head * DHsz + dh) * Tsz + t] = (__bf16)v;
                }
            }
        }
    }
}

// ---------------------------------------------------------------------------
// Kernel 3: flash attention per (b,h).  Block = 128 query rows (8 waves x 16);
// 64-key K/Vt tiles async-staged, double-buffered, shared by all waves.
// ---------------------------------------------------------------------------
__device__ __forceinline__ void stage_attn64(const __bf16* __restrict__ Kh,
                                             const __bf16* __restrict__ Vt,
                                             int key0, __bf16* kbuf, __bf16* vbuf) {
    int tid = threadIdx.x;
    int row = tid >> 2;
    int c   = (tid & 3) * 16;              // 32B chunk of a 128B row
    {   // K tile: 64 keys x 64 dh (8KB)
        unsigned int l = lds_addr_of(kbuf + row * 64 + c);
        const __bf16* gp = Kh + (size_t)(key0 + row) * DHsz + c;
        async_b128(l,      gp);
        async_b128(l + 16, gp + 8);
    }
    {   // Vt tile: 64 dh x 64 keys (8KB)
        unsigned int l = lds_addr_of(vbuf + row * 64 + c);
        const __bf16* gp = Vt + (size_t)row * Tsz + key0 + c;
        async_b128(l,      gp);
        async_b128(l + 16, gp + 8);
    }
}

__global__ void attn_kernel(const __bf16* __restrict__ qb, const __bf16* __restrict__ kb,
                            const __bf16* __restrict__ vtb, __bf16* __restrict__ ao) {
    __shared__ __align__(16) __bf16 Kbuf[2][64 * 64];   // 16KB
    __shared__ __align__(16) __bf16 Vbuf[2][64 * 64];   // 16KB
    __shared__ __align__(16) __bf16 Plds[8][16 * 64];   // 16KB

    int bh   = blockIdx.x >> 3;
    int qblk = blockIdx.x & 7;
    int wave = threadIdx.x >> 5;
    int lane = threadIdx.x & 31;
    int hf = lane >> 4, col = lane & 15;
    int q0 = qblk * 128 + wave * 16;

    const __bf16* Qh = qb  + (size_t)bh * Tsz * DHsz;
    const __bf16* Kh = kb  + (size_t)bh * Tsz * DHsz;
    const __bf16* Vt = vtb + (size_t)bh * DHsz * Tsz;
    __bf16* pbuf = &Plds[wave][0];

    v16bf aq0 = load_frag_a(Qh, DHsz, q0, 0);
    v16bf aq1 = load_frag_a(Qh, DHsz, q0, 32);

    v8f o[4] = {};
    float mr[8], lr[8];
#pragma unroll
    for (int r = 0; r < 8; ++r) { mr[r] = -1.0e30f; lr[r] = 0.0f; }

    stage_attn64(Kh, Vt, 0,  Kbuf[0], Vbuf[0]);
    stage_attn64(Kh, Vt, 64, Kbuf[1], Vbuf[1]);

    for (int it = 0; it < 16; ++it) {
        ASYNC_WAIT_4();
        __syncthreads();
        const __bf16* Kc = Kbuf[it & 1];
        const __bf16* Vc = Vbuf[it & 1];

        // S = Q * K^T : four 16x16 tiles over this 64-key block (exp2 domain)
        v8f st[4];
#pragma unroll
        for (int j = 0; j < 4; ++j) {
            v8f s = {};
            s = wmma_bf16(aq0, load_frag_bt(Kc, 64, j * 16, 0),  s);
            s = wmma_bf16(aq1, load_frag_bt(Kc, 64, j * 16, 32), s);
            st[j] = s;
        }

        float sc[8];
#pragma unroll
        for (int r = 0; r < 8; ++r) {
            float mx = fmaxf(fmaxf(st[0][r], st[1][r]), fmaxf(st[2][r], st[3][r]));
            mx = fmaxf(mx, __shfl_xor(mx, 1));
            mx = fmaxf(mx, __shfl_xor(mx, 2));
            mx = fmaxf(mx, __shfl_xor(mx, 4));
            mx = fmaxf(mx, __shfl_xor(mx, 8));
            float mnew = fmaxf(mr[r], mx);
            sc[r] = __builtin_amdgcn_exp2f(mr[r] - mnew);
            mr[r] = mnew;
            float rs = 0.0f;
#pragma unroll
            for (int j = 0; j < 4; ++j) {
                float p = __builtin_amdgcn_exp2f(st[j][r] - mnew);
                st[j][r] = p;              // P overwrites S in place
                rs += p;
            }
            rs += __shfl_xor(rs, 1);
            rs += __shfl_xor(rs, 2);
            rs += __shfl_xor(rs, 4);
            rs += __shfl_xor(rs, 8);
            lr[r] = lr[r] * sc[r] + rs;
        }
#pragma unroll
        for (int nt = 0; nt < 4; ++nt)
#pragma unroll
            for (int r = 0; r < 8; ++r) o[nt][r] *= sc[r];

        // P: C-layout -> LDS [16 rows][64 keys]
#pragma unroll
        for (int j = 0; j < 4; ++j)
#pragma unroll
            for (int r = 0; r < 8; ++r)
                pbuf[(r + 8 * hf) * 64 + j * 16 + col] = (__bf16)st[j][r];
        asm volatile("s_wait_dscnt 0x0" ::: "memory");

        v16bf ap0 = load_frag_a(pbuf, 64, 0, 0);
        v16bf ap1 = load_frag_a(pbuf, 64, 0, 32);
#pragma unroll
        for (int nt = 0; nt < 4; ++nt) {
            o[nt] = wmma_bf16(ap0, load_frag_bt(Vc, 64, nt * 16, 0),  o[nt]);
            o[nt] = wmma_bf16(ap1, load_frag_bt(Vc, 64, nt * 16, 32), o[nt]);
        }
        asm volatile("" ::: "memory");

        __syncthreads();
        if (it + 2 < 16)
            stage_attn64(Kh, Vt, (it + 2) * 64, Kbuf[it & 1], Vbuf[it & 1]);
    }

    // Normalize; write bf16 [B,T,D] at column h*64+dh for the proj GEMM.
    int b = bh / Hsz;
    int h = bh - b * Hsz;
#pragma unroll
    for (int r = 0; r < 8; ++r) {
        float inv = 1.0f / lr[r];
        int row = q0 + r + 8 * hf;
        size_t base = ((size_t)(b * Tsz + row)) * Dsz + h * DHsz;
#pragma unroll
        for (int nt = 0; nt < 4; ++nt)
            ao[base + nt * 16 + col] = (__bf16)(o[nt][r] * inv);
    }
}

// ---------------------------------------------------------------------------
// Kernel 4: output projection. out[m,e] = sum_d ao[m,d] * Wp[e,d] + b[e] (f32)
// Grid: 64 x 6 = 384 blocks.
// ---------------------------------------------------------------------------
__global__ void proj_gemm(const __bf16* __restrict__ ao, const __bf16* __restrict__ wp,
                          const float* __restrict__ bias, float* __restrict__ out) {
    __shared__ __align__(16) __bf16 Ab[2 * 8192];
    __shared__ __align__(16) __bf16 Bb[2 * 8192];
    int mb = blockIdx.x & 63;
    int nb = blockIdx.x >> 6;
    int m0_blk = mb * 128, n0_blk = nb * 128;
    int wave = threadIdx.x >> 5;
    int wm = (wave >> 1) * 32;
    int wn = (wave & 1) * 64;
    int lane = threadIdx.x & 31;
    int hf = lane >> 4, col = lane & 15;

    v8f acc[2][4] = {};
    gemm_mainloop_128(ao, wp, m0_blk, n0_blk, Ab, Bb, wm, wn, acc);

#pragma unroll
    for (int nt = 0; nt < 4; ++nt) {
        int e = n0_blk + wn + nt * 16 + col;
        float bv = bias[e];
#pragma unroll
        for (int mt = 0; mt < 2; ++mt) {
#pragma unroll
            for (int r = 0; r < 8; ++r) {
                int gm = m0_blk + wm + mt * 16 + r + 8 * hf;
                out[(size_t)gm * Dsz + e] = acc[mt][nt][r] + bv;
            }
        }
    }
}

// ---------------------------------------------------------------------------
// Host launcher
// ---------------------------------------------------------------------------
extern "C" void kernel_launch(void* const* d_in, const int* in_sizes, int n_in,
                              void* d_out, int out_size, void* d_ws, size_t ws_size,
                              hipStream_t stream) {
    const float* x      = (const float*)d_in[0];
    const float* qkv_w  = (const float*)d_in[1];
    const float* qkv_b  = (const float*)d_in[2];
    const float* proj_w = (const float*)d_in[3];
    const float* proj_b = (const float*)d_in[4];

    char* ws = (char*)d_ws;
    size_t off = 0;
    __bf16* xb  = (__bf16*)(ws + off); off += (size_t)Msz * Dsz * 2;
    __bf16* wq  = (__bf16*)(ws + off); off += (size_t)D3sz * Dsz * 2;
    __bf16* wp  = (__bf16*)(ws + off); off += (size_t)Dsz * Dsz * 2;
    __bf16* qb  = (__bf16*)(ws + off); off += (size_t)Msz * Dsz * 2;
    __bf16* kb  = (__bf16*)(ws + off); off += (size_t)Msz * Dsz * 2;
    __bf16* vtb = (__bf16*)(ws + off); off += (size_t)Msz * Dsz * 2;
    __bf16* ao  = (__bf16*)(ws + off); off += (size_t)Msz * Dsz * 2;

    cvt_f32_to_bf16<<<4096, 256, 0, stream>>>(x,      xb, Msz * Dsz);
    cvt_f32_to_bf16<<<2048, 256, 0, stream>>>(qkv_w,  wq, D3sz * Dsz);
    cvt_f32_to_bf16<<<1024, 256, 0, stream>>>(proj_w, wp, Dsz * Dsz);

    qkv_gemm<<<1152, 256, 0, stream>>>(xb, wq, qkv_b, qb, kb, vtb);
    attn_kernel<<<768, 256, 0, stream>>>(qb, kb, vtb, ao);
    proj_gemm<<<384, 256, 0, stream>>>(ao, wp, proj_b, (float*)d_out);
}
